// MultiModalInputEmbeddings_25984552141602
// MI455X (gfx1250) — matrix operand order repaired and checked by reading, not compiled, not run on hardware
//
#include <hip/hip_runtime.h>
#include <hip/hip_bf16.h>
#include <stdint.h>

typedef __attribute__((ext_vector_type(16))) __bf16 v16bf;
typedef __attribute__((ext_vector_type(8)))  float  v8f;
typedef __attribute__((ext_vector_type(4)))  unsigned int du32x4;
typedef __attribute__((ext_vector_type(8)))  int         di32x8;
typedef __attribute__((ext_vector_type(4)))  int         di32x4;

#define TOKENS 32768   // B*S = 64*512
#define FPD    768
#define HIDD   768
#define FFN    3072
#define TPW    16      // tokens per workgroup
#define CHUNK  128     // hidden units per chunk (8 waves x 16 cols)

__device__ __forceinline__ unsigned short f2bf(float f) {
  unsigned u = __float_as_uint(f);
  u += 0x7FFFu + ((u >> 16) & 1u);   // round-to-nearest-even
  return (unsigned short)(u >> 16);
}

__device__ __forceinline__ unsigned pack_bf2(float a, float b) {
#if __has_builtin(__builtin_amdgcn_cvt_pk_bf16_f32)
  auto p = __builtin_amdgcn_cvt_pk_bf16_f32(a, b);
  unsigned r; __builtin_memcpy(&r, &p, sizeof(r));
  return r;
#else
  return (unsigned)f2bf(a) | ((unsigned)f2bf(b) << 16);
#endif
}

__device__ __forceinline__ unsigned short f2bf_fast(float a) {
#if __has_builtin(__builtin_amdgcn_cvt_pk_bf16_f32)
  return (unsigned short)(pack_bf2(a, 0.f) & 0xFFFFu);
#else
  return f2bf(a);
#endif
}

union ABu { v16bf v; uint4 u[2]; };

// ---- pre-pass: fc1_w [768,3072] -> W1T [3072][768] bf16; fc2_w [3072,768] -> W2T [768][3072] bf16
__global__ void conv_w(const float* __restrict__ W1, const float* __restrict__ W2,
                       unsigned short* __restrict__ W1T, unsigned short* __restrict__ W2T) {
  int idx = blockIdx.x * blockDim.x + threadIdx.x;
  const int N1 = FPD * FFN;
  if (idx < N1) {
    int n = idx / FPD, k = idx - n * FPD;
    W1T[idx] = f2bf_fast(W1[(size_t)k * FFN + n]);
  } else {
    int j = idx - N1;
    if (j < FFN * HIDD) {
      int n = j / FFN, k = j - n * FFN;
      W2T[j] = f2bf_fast(W2[(size_t)k * HIDD + n]);
    }
  }
}

// ---- pre-pass: SMILES_fps f32 -> bf16 (row-major, same layout) ----
__global__ void conv_x(const float* __restrict__ fps, unsigned int* __restrict__ Xb) {
  int idx = blockIdx.x * blockDim.x + threadIdx.x;  // one f32 pair per thread
  float2 f = ((const float2*)fps)[idx];
  Xb[idx] = pack_bf2(f.x, f.y);
}

__global__ void __launch_bounds__(256)
mm_embed(const unsigned short* __restrict__ Xbf,
         const int*   __restrict__ word_tok,
         const float* __restrict__ values,
         const int*   __restrict__ ttype,
         const int*   __restrict__ posid,
         const float* __restrict__ b1,
         const float* __restrict__ b2,
         const float* __restrict__ prop_emb,
         const float* __restrict__ val_w,
         const float* __restrict__ val_b,
         const float* __restrict__ pos_emb,
         const float* __restrict__ type_emb,
         const float* __restrict__ ln_g,
         const float* __restrict__ ln_b,
         const unsigned short* __restrict__ W1T,
         const unsigned short* __restrict__ W2T,
         float* __restrict__ out)
{
  // LDS layout (Es overlaps Xs/Hs: only live after the GEMM loop)
  __shared__ __align__(16) char smem[49152 + 2048 + 128];
  unsigned short* Xs = (unsigned short*)smem;              // 16 x 768 bf16  (24576 B)
  unsigned short* Hs = (unsigned short*)(smem + 24576);    // 16 x 128 bf16  ( 4096 B)
  float* Es    = (float*)smem;                             // 16 x 768 f32   (49152 B)
  float* red   = (float*)(smem + 49152);                   // 2 x 256 f32
  float* stats = (float*)(smem + 51200);                   // 2 x 16  f32

  const int tid   = threadIdx.x;
  const int wave  = tid >> 5;
  const int lane  = tid & 31;
  const int ln16  = lane & 15;
  const int lh    = lane >> 4;
  const int token0 = blockIdx.x * TPW;

  // ---- stage X tile (16 x 768 bf16, 24 KB) into LDS via Tensor Data Mover ----
#if __has_builtin(__builtin_amdgcn_tensor_load_to_lds)
  if (wave == 0) {
    unsigned long long ga = (unsigned long long)(const void*)(Xbf + (size_t)token0 * FPD);
    unsigned ldsa = (unsigned)(unsigned long long)(void*)Xs;
    du32x4 g0 = { 1u,                                  // count=1, no gather
                  ldsa,                                // lds_addr
                  (unsigned)(ga & 0xFFFFFFFFu),        // global_addr[31:0]
                  (unsigned)((ga >> 32) & 0x01FFFFFFu) | (2u << 30) };  // addr[56:32] | type=2
    di32x8 g1 = { (int)0x00010000,        // wg_mask=0, data_size=1 (2 bytes)
                  (int)(768u << 16),      // tensor_dim0 = 768
                  (int)0x80000000u,       // tensor_dim1 = 32768
                  (int)(768u << 16),      // tile_dim0 = 768
                  16,                     // tile_dim1 = 16 rows
                  768,                    // tensor_dim0_stride = 768
                  0, 0 };
    di32x4 gz = { 0, 0, 0, 0 };
#if __clang_major__ >= 23
    di32x8 gz8 = { 0, 0, 0, 0, 0, 0, 0, 0 };
    __builtin_amdgcn_tensor_load_to_lds(g0, g1, gz, gz, gz8, 0);
#else
    __builtin_amdgcn_tensor_load_to_lds(g0, g1, gz, gz, 0);
#endif
    __builtin_amdgcn_s_wait_tensorcnt(0);
  }
#else
  {  // fallback: cooperative plain copy (Xbf already bf16)
    const uint4* gx = (const uint4*)(Xbf + (size_t)token0 * FPD);
    uint4* xs4 = (uint4*)Xs;
    #pragma unroll
    for (int i = 0; i < (TPW * FPD * 2 / 16) / 256; ++i)
      xs4[tid + i * 256] = gx[tid + i * 256];
  }
#endif
  __syncthreads();

  v8f acc[6];
  #pragma unroll
  for (int j = 0; j < 6; ++j) acc[j] = {};

  const unsigned short* arow = &Xs[ln16 * FPD];    // A row for GEMM1 (M = ln16)
  const unsigned short* hrow = &Hs[ln16 * CHUNK];  // A row for GEMM2

  for (int kb = 0; kb < FFN; kb += CHUNK) {
    // ---- phase 1: h[16, CHUNK] = relu(X @ W1[:, kb:kb+CHUNK] + b1), wave owns 16 cols ----
    {
      const int colg = kb + wave * 16;
      const unsigned short* brow = W1T + (size_t)(colg + ln16) * FPD;
      v8f a1 = {};
      #pragma unroll 4
      for (int k = 0; k < FPD; k += 32) {
        ABu a, b;
        a.u[0] = *(const uint4*)&arow[k + 8 * lh];        // K = k + 8h + 0..7
        a.u[1] = *(const uint4*)&arow[k + 16 + 8 * lh];   // K = k + 16 + 8h + 0..7
        b.u[0] = *(const uint4*)&brow[k + 8 * lh];        // K = k + 8h + 0..7
        b.u[1] = *(const uint4*)&brow[k + 16 + 8 * lh];   // K = k + 16 + 8h + 0..7
        a1 = __builtin_amdgcn_wmma_f32_16x16x32_bf16(false, a.v, false, b.v,
                                                     (short)0, a1, false, false);
      }
      const float bias = b1[colg + ln16];
      #pragma unroll
      for (int r = 0; r < 8; ++r) {
        float h = a1[r] + bias;
        h = h > 0.f ? h : 0.f;
        Hs[(r + 8 * lh) * CHUNK + wave * 16 + ln16] = f2bf_fast(h);
      }
    }
    __syncthreads();

    // ---- phase 2: out[16,768] += h @ W2[kb:kb+CHUNK,:]; A-frag hoisted across 6 tiles ----
    #pragma unroll
    for (int k = 0; k < CHUNK; k += 32) {
      ABu a;
      a.u[0] = *(const uint4*)&hrow[k + 8 * lh];
      a.u[1] = *(const uint4*)&hrow[k + 16 + 8 * lh];
      #pragma unroll
      for (int j = 0; j < 6; ++j) {
        const unsigned short* brow = W2T + (size_t)(wave * 96 + j * 16 + ln16) * FFN + kb;
        ABu b;
        b.u[0] = *(const uint4*)&brow[k + 8 * lh];
        b.u[1] = *(const uint4*)&brow[k + 16 + 8 * lh];
        acc[j] = __builtin_amdgcn_wmma_f32_16x16x32_bf16(false, a.v, false, b.v,
                                                         (short)0, acc[j], false, false);
      }
    }
    __syncthreads();
  }

  // ---- spill SMILES-MLP result to LDS; D layout: M = r + 8h, N = ln16 ----
  #pragma unroll
  for (int j = 0; j < 6; ++j) {
    const int col = wave * 96 + j * 16 + ln16;
    #pragma unroll
    for (int r = 0; r < 8; ++r)
      Es[(r + 8 * lh) * HIDD + col] = acc[j][r];
  }
  __syncthreads();

  // ---- epilogue: branch select + pos/type add + LayerNorm; 16 lanes per token ----
  const int tloc = tid >> 4;
  const int c0   = tid & 15;
  const int gt   = token0 + tloc;
  const int tt   = ttype[gt];
  const int wtk  = word_tok[gt];
  const float vv = values[gt];
  const int pp   = posid[gt];

  float s = 0.f, q = 0.f;
  for (int c = c0; c < HIDD; c += 16) {
    float base;
    if (tt == 0)      base = prop_emb[(size_t)wtk * HIDD + c];
    else if (tt == 1) base = Es[tloc * HIDD + c] + b2[c];
    else if (tt == 2) base = vv * val_w[c] + val_b[c];
    else              base = 0.f;  // special_mask in reference is identically false
    float e = base + pos_emb[(size_t)pp * HIDD + c] + type_emb[(size_t)tt * HIDD + c];
    Es[tloc * HIDD + c] = e;
    s += e; q += e * e;
  }
  red[tloc * 16 + c0]       = s;
  red[256 + tloc * 16 + c0] = q;
  __syncthreads();
  if (tid < 16) {
    float ss = 0.f, qq = 0.f;
    #pragma unroll
    for (int i = 0; i < 16; ++i) { ss += red[tid * 16 + i]; qq += red[256 + tid * 16 + i]; }
    float mu  = ss * (1.0f / 768.0f);
    float var = qq * (1.0f / 768.0f) - mu * mu;
    stats[tid]      = mu;
    stats[16 + tid] = rsqrtf(var + 1e-12f);
  }
  __syncthreads();
  const float mu = stats[tloc], rv = stats[16 + tloc];
  float* orow = out + (size_t)gt * HIDD;
  for (int c = c0; c < HIDD; c += 16)
    orow[c] = (Es[tloc * HIDD + c] - mu) * rv * ln_g[c] + ln_b[c];
}

extern "C" void kernel_launch(void* const* d_in, const int* in_sizes, int n_in,
                              void* d_out, int out_size, void* d_ws, size_t ws_size,
                              hipStream_t stream) {
  (void)in_sizes; (void)n_in; (void)out_size; (void)ws_size;
  const float* fps   = (const float*)d_in[2];
  const int*   wtok  = (const int*)d_in[3];
  const float* vals  = (const float*)d_in[4];
  const int*   tt    = (const int*)d_in[5];
  const int*   pos   = (const int*)d_in[6];
  const float* W1    = (const float*)d_in[7];
  const float* b1    = (const float*)d_in[8];
  const float* W2    = (const float*)d_in[9];
  const float* b2    = (const float*)d_in[10];
  const float* prop  = (const float*)d_in[11];
  const float* valw  = (const float*)d_in[12];
  const float* valb  = (const float*)d_in[13];
  const float* pose  = (const float*)d_in[14];
  const float* typee = (const float*)d_in[15];
  const float* lng   = (const float*)d_in[16];
  const float* lnb   = (const float*)d_in[17];

  unsigned short* W1T = (unsigned short*)d_ws;                 // [3072][768] bf16
  unsigned short* W2T = W1T + (size_t)FPD * FFN;               // [768][3072] bf16
  unsigned short* Xbf = W2T + (size_t)FFN * HIDD;              // [32768][768] bf16 (48 MB)

  const int Nconv = 2 * FPD * FFN;
  conv_w<<<(Nconv + 255) / 256, 256, 0, stream>>>(W1, W2, W1T, W2T);
  const int Npair = TOKENS * FPD / 2;
  conv_x<<<Npair / 256, 256, 0, stream>>>(fps, (unsigned int*)Xbf);
  mm_embed<<<TOKENS / TPW, 256, 0, stream>>>(Xbf, wtok, vals, tt, pos, b1, b2, prop,
                                             valw, valb, pose, typee, lng, lnb,
                                             W1T, W2T, (float*)d_out);
}
